// QuantizedDot_46789373723182
// MI455X (gfx1250) — 8-bit matrix operand order repaired and checked by reading, compiled, not run on hardware
//
#include <hip/hip_runtime.h>


typedef int v8i __attribute__((ext_vector_type(8)));
typedef unsigned u32x4 __attribute__((ext_vector_type(4)));
typedef int i32x4v __attribute__((ext_vector_type(4)));
typedef int i32x8v __attribute__((ext_vector_type(8)));

#define K_DIM 512
#define N_DIM 512
#define M_DIM 65536
#define BM 128
#define BN 128
#define CLIPV 127.0f
#define EPSV 1e-6f

#if __has_builtin(__builtin_amdgcn_tensor_load_to_lds) && \
    __has_builtin(__builtin_amdgcn_s_wait_tensorcnt)
#define USE_TDM 1
#else
#define USE_TDM 0
#endif

__global__ void init_bound_kernel(unsigned* __restrict__ bound) {
  bound[threadIdx.x] = 0u;
}

__global__ __launch_bounds__(256) void act_absmax_kernel(
    const float* __restrict__ act, unsigned* __restrict__ bound) {
  const int t = threadIdx.x;
  const size_t row0 = (size_t)blockIdx.x * 256;
  const float* p = act + row0 * K_DIM;
  float m0 = 0.f, m1 = 0.f;
  for (int r = 0; r < 256; ++r) {
    m0 = fmaxf(m0, fabsf(p[(size_t)r * K_DIM + t]));
    m1 = fmaxf(m1, fabsf(p[(size_t)r * K_DIM + t + 256]));
  }
  atomicMax(&bound[t],       __float_as_uint(m0));
  atomicMax(&bound[t + 256], __float_as_uint(m1));
}

__global__ __launch_bounds__(512) void wq_kernel(
    const float* __restrict__ w, const unsigned* __restrict__ bound,
    float* __restrict__ act_scale_g, float* __restrict__ inv_w_scale_g,
    int* __restrict__ wq_t) {
  __shared__ float as_lds[K_DIM];
  const int n = threadIdx.x;
  const float b  = __uint_as_float(bound[n]);
  const float as = CLIPV / fmaxf(b, EPSV);
  act_scale_g[n] = as;
  as_lds[n] = as;
  __syncthreads();

  float wmax = 0.f;
  for (int k = 0; k < K_DIM; ++k) {
    const float wsv = w[(size_t)k * N_DIM + n] / as_lds[k];
    wmax = fmaxf(wmax, fabsf(wsv));
  }
  const float wsc = CLIPV / fmaxf(wmax, EPSV);
  inv_w_scale_g[n] = 1.0f / wsc;

  for (int k = 0; k < K_DIM; k += 4) {
    int packed = 0;
#pragma unroll
    for (int e = 0; e < 4; ++e) {
      const float wsv = w[(size_t)(k + e) * N_DIM + n] / as_lds[k + e];
      const float q = rintf(fminf(fmaxf(wsv * wsc, -CLIPV), CLIPV));
      packed |= (((int)q) & 255) << (8 * e);
    }
    wq_t[n * (K_DIM / 4) + (k >> 2)] = packed;
  }
}

__device__ __forceinline__ int quant_pack4(float4 v, float4 s) {
  const float q0 = rintf(fminf(fmaxf(v.x * s.x, -CLIPV), CLIPV));
  const float q1 = rintf(fminf(fmaxf(v.y * s.y, -CLIPV), CLIPV));
  const float q2 = rintf(fminf(fmaxf(v.z * s.z, -CLIPV), CLIPV));
  const float q3 = rintf(fminf(fmaxf(v.w * s.w, -CLIPV), CLIPV));
  return (((int)q0) & 255) | ((((int)q1) & 255) << 8) |
         ((((int)q2) & 255) << 16) | ((((int)q3) & 255) << 24);
}

__global__ __launch_bounds__(256) void act_quant_kernel(
    const float* __restrict__ act, const float* __restrict__ act_scale_g,
    int4* __restrict__ aq4) {
  __shared__ __align__(16) float as_lds[K_DIM];
  const int tid = threadIdx.x;
  as_lds[tid]       = act_scale_g[tid];
  as_lds[tid + 256] = act_scale_g[tid + 256];
  __syncthreads();

  const int lane = tid & 31;
  const int wv = tid >> 5;
  const int l16 = lane & 15;
  const int lhi = lane >> 4;
  const int khA = lhi * 16;

  const size_t mt = (size_t)blockIdx.x * 8 + wv;
  const float* arow = act + (mt * 16 + (size_t)l16) * K_DIM;

#pragma unroll
  for (int kt = 0; kt < 8; ++kt) {
    const int kc = kt * 64;
    v8i a;
#pragma unroll
    for (int r = 0; r < 8; ++r) {
      const int kb = kc + ((r >> 2) << 5) + ((r & 3) << 2) + khA;
      const float4 s = *(const float4*)&as_lds[kb];
      const float4 v = *(const float4*)(arow + kb);
      a[r] = quant_pack4(v, s);
    }
    const size_t toff = (mt * 8 + (size_t)kt) * 64;
    aq4[toff + lane]      = make_int4(a[0], a[1], a[2], a[3]);
    aq4[toff + 32 + lane] = make_int4(a[4], a[5], a[6], a[7]);
  }
}

__global__ __launch_bounds__(256) void gemm_kernel(
    const int4* __restrict__ aq4, const int* __restrict__ wq_t,
    const float* __restrict__ inv_w_scale_g, float* __restrict__ out) {
  __shared__ __align__(16) int Bq[BN * (K_DIM / 4)];

  const int tid = threadIdx.x;
  const int nbase = blockIdx.x * BN;

#if USE_TDM
  if (tid < 32) {
    const unsigned long long gaddr =
        (unsigned long long)(const char*)wq_t + (unsigned long long)nbase * K_DIM;
    const unsigned lds_off = (unsigned)(unsigned long long)(const void*)&Bq[0];
    const u32x4 g0 = {1u, lds_off, (unsigned)(gaddr & 0xffffffffu),
                      (unsigned)((gaddr >> 32) & 0x01ffffffu) | (2u << 30)};
    const i32x8v g1 = {0,
                       (int)(512u << 16),
                       (int)(512u << 16),
                       (int)(512u << 16),
                       128,
                       512,
                       0, 0};
    const i32x4v gz4 = {0, 0, 0, 0};
#if __clang_major__ >= 23
    const i32x8v gz8 = {0, 0, 0, 0, 0, 0, 0, 0};
    __builtin_amdgcn_tensor_load_to_lds(g0, g1, gz4, gz4, gz8, 0);
#else
    __builtin_amdgcn_tensor_load_to_lds(g0, g1, gz4, gz4, 0);
#endif
    __builtin_amdgcn_s_wait_tensorcnt(0);
  }
#else
  {
    const int4* src = (const int4*)(wq_t + (size_t)nbase * (K_DIM / 4));
    int4* dst = (int4*)Bq;
#pragma unroll
    for (int i = 0; i < 16; ++i) dst[tid + i * 256] = src[tid + i * 256];
  }
#endif
  __syncthreads();

  const int lane = tid & 31;
  const int wv = tid >> 5;
  const int wm = wv & 3;
  const int wn = wv >> 2;
  const int l16 = lane & 15;
  const int lhi = lane >> 4;

  const size_t mt0 = (size_t)blockIdx.y * 8 + (size_t)(wm * 2);

  v8i acc[2][4];
  {
    const v8i z = {0, 0, 0, 0, 0, 0, 0, 0};
#pragma unroll
    for (int i = 0; i < 2; ++i)
#pragma unroll
      for (int j = 0; j < 4; ++j) acc[i][j] = z;
  }

  for (int kt = 0; kt < 8; ++kt) {
    const int kc = kt * 64;
    v8i a[2];
#pragma unroll
    for (int i = 0; i < 2; ++i) {
      const size_t toff = ((mt0 + (size_t)i) * 8 + (size_t)kt) * 64;
      const int4 lo = aq4[toff + lane];
      const int4 hi = aq4[toff + 32 + lane];
      const v8i t = {lo.x, lo.y, lo.z, lo.w, hi.x, hi.y, hi.z, hi.w};
      a[i] = t;
      __builtin_prefetch(&aq4[toff + 64 + lane], 0, 3);
    }
    v8i bf[4];
#pragma unroll
    for (int j = 0; j < 4; ++j) {
      const char* bp = (const char*)Bq +
                       ((wn * 64 + j * 16 + l16) * K_DIM) + kc + lhi * 16;
      const int4 lo = *(const int4*)bp;
      const int4 hi = *(const int4*)(bp + 32);
      const v8i t = {lo.x, lo.y, lo.z, lo.w, hi.x, hi.y, hi.z, hi.w};
      bf[j] = t;
    }
#pragma unroll
    for (int j = 0; j < 4; ++j) {
      acc[0][j] = __builtin_amdgcn_wmma_i32_16x16x64_iu8(
          true, a[0], true, bf[j], acc[0][j], false, false);
      acc[1][j] = __builtin_amdgcn_wmma_i32_16x16x64_iu8(
          true, a[1], true, bf[j], acc[1][j], false, false);
    }
  }

#pragma unroll
  for (int j = 0; j < 4; ++j) {
    const int ncol = nbase + wn * 64 + j * 16 + l16;
    const float iws = inv_w_scale_g[ncol];
#pragma unroll
    for (int i = 0; i < 2; ++i) {
      const size_t mr = mt0 * 16 + (size_t)(i * 16 + lhi * 8);
#pragma unroll
      for (int r = 0; r < 8; ++r) {
        __builtin_nontemporal_store((float)acc[i][j][r] * iws,
                                    &out[(mr + (size_t)r) * N_DIM + ncol]);
      }
    }
  }
}

extern "C" void kernel_launch(void* const* d_in, const int* in_sizes, int n_in,
                              void* d_out, int out_size, void* d_ws,
                              size_t ws_size, hipStream_t stream) {
  const float* act = (const float*)d_in[0];
  const float* w   = (const float*)d_in[1];
  float* out = (float*)d_out;

  char* ws = (char*)d_ws;
  int4*     aq4          = (int4*)ws;
  unsigned* bound        = (unsigned*)(ws + 33554432);
  float*    act_scale_g  = (float*)(ws + 33554432 + 2048);
  float*    inv_w_scale  = (float*)(ws + 33554432 + 4096);
  int*      wq_t         = (int*)(ws + 33554432 + 6144);

  init_bound_kernel<<<1, 512, 0, stream>>>(bound);
  act_absmax_kernel<<<M_DIM / 256, 256, 0, stream>>>(act, bound);
  wq_kernel<<<1, 512, 0, stream>>>(w, bound, act_scale_g, inv_w_scale, wq_t);
  act_quant_kernel<<<M_DIM / BM, 256, 0, stream>>>(act, act_scale_g, aq4);

  dim3 grid(N_DIM / BN, M_DIM / BM);
  gemm_kernel<<<grid, 256, 0, stream>>>(aq4, wq_t, inv_w_scale, out);
}
